// EvoformerIteration_9268539425325
// MI455X (gfx1250) — matrix operand order repaired and checked by reading, not compiled, not run on hardware
//
#include <hip/hip_runtime.h>
#include <hip/hip_bf16.h>
#include <math.h>

// ---------------------------------------------------------------------------
// Evoformer iteration for MI455X (gfx1250, wave32, WMMA).
// All contractions: v_wmma_f32_16x16x32_bf16, f32 accumulate.
// One generic strided/batched GEMM kernel, 2x2 WMMA tiles per wave (32x32),
// division-free addressing (per-32k pointer steps), float4 fragment loads
// when strides permit, next-tile global_prefetch_b8. Epilogue fuses
// scale + bias + act + elementwise-mul + residual.
// ---------------------------------------------------------------------------

typedef __attribute__((ext_vector_type(16))) __bf16 bf16x16;
typedef __attribute__((ext_vector_type(8)))  float  f32x8;

struct GemmParams {
  const float* A; const float* B; float* C;
  const float* bias; const float* emul; const float* resid;
  long long asr, ak0, astep, ab0, ab1;   // A: row stride, k-elem stride, per-32k step, batch0/1
  long long bk0, bn0, bstep, bb0, bb1;   // B: k-elem stride, col stride, per-32k step, batch0/1
  long long csr, cn0, cb0, cb1;          // C (emul/resid share these)
  long long gsr, gsc, gb0, gb1;          // bias strides
  int M, N, K, nbs, act;                 // nbs: batch-split shift; act: 0 none,1 relu,2 sigmoid
  float scale, biasScale;
};

__device__ __forceinline__ f32x8 wmma_bf16(const bf16x16& a, const bf16x16& b, const f32x8& c) {
  return __builtin_amdgcn_wmma_f32_16x16x32_bf16(false, a, false, b, (short)0, c, false, false);
}

template<bool NB>
__device__ __forceinline__ void store_tile(const GemmParams& p, float* C, const float* E,
                                           const float* R, const float* biasB,
                                           const f32x8& a, int mb, int n, int half) {
  if (NB && n >= p.N) return;
  const long long ncol = (long long)n * p.cn0;
  const float* bp = biasB ? biasB + (long long)n * p.gsc : nullptr;
#pragma unroll
  for (int r = 0; r < 8; ++r) {
    const int m = mb + r + half * 8;
    float v = a[r] * p.scale;
    if (bp) v += bp[(long long)m * p.gsr] * p.biasScale;
    if (p.act == 1)      v = fmaxf(v, 0.f);
    else if (p.act == 2) v = 1.f / (1.f + __expf(-v));
    const long long idx = (long long)m * p.csr + ncol;
    float t = v;
    if (E) t *= E[idx];
    if (R) t += R[idx];
    C[idx] = t;
  }
}

template<bool AC, bool BC, bool NB>
__global__ void wmma_gemm(GemmParams p) {
  const int lane = threadIdx.x & 31;
  const int half = lane >> 4;            // 0: lanes 0-15, 1: lanes 16-31
  const int l15  = lane & 15;
  const int n0 = blockIdx.x * 32;
  const int m0 = blockIdx.y * 32;
  const int b  = blockIdx.z;
  const int b0 = b >> p.nbs, b1 = b & ((1 << p.nbs) - 1);

  const float* Abase = p.A + (long long)b0 * p.ab0 + (long long)b1 * p.ab1;
  const float* Bbase = p.B + (long long)b0 * p.bb0 + (long long)b1 * p.bb1;

  // A: lane carries row; two contiguous 8-elem k-runs at half*8 and 16+half*8
  const float* Ap0 = Abase + (long long)(m0 + l15) * p.asr;
  const float* Ap1 = Ap0 + 16LL * p.asr;
  // B: lane carries col; two contiguous 8-elem k-runs at half*8 and 16+half*8
  const int nB0 = n0 + l15, nB1 = n0 + 16 + l15;
  const int nc0 = NB ? min(nB0, p.N - 1) : nB0;
  const int nc1 = NB ? min(nB1, p.N - 1) : nB1;
  const float* Bp0 = Bbase + (long long)nc0 * p.bn0;
  const float* Bp1 = Bbase + (long long)nc1 * p.bn0;

  f32x8 acc00 = {0.f,0.f,0.f,0.f,0.f,0.f,0.f,0.f};
  f32x8 acc01 = acc00, acc10 = acc00, acc11 = acc00;

  const int nk = p.K >> 5;
  for (int t = 0; t < nk; ++t) {
    // Prefetch next 32-k tile (speculative; dropped silently past the end).
    __builtin_prefetch(Ap0 + p.astep, 0, 1);
    __builtin_prefetch(Ap1 + p.astep, 0, 1);
    __builtin_prefetch(Bp0 + p.bstep, 0, 1);
    __builtin_prefetch(Bp1 + p.bstep, 0, 1);

    float a0[16], a1[16], v0[16], v1[16];
    if (AC) {
      const float* s0 = Ap0 + half * 8;
      const float* s1 = Ap1 + half * 8;
      float4 x0 = *(const float4*)(s0);      float4 x1 = *(const float4*)(s0 + 4);
      float4 x2 = *(const float4*)(s0 + 16); float4 x3 = *(const float4*)(s0 + 20);
      float4 y0 = *(const float4*)(s1);      float4 y1 = *(const float4*)(s1 + 4);
      float4 y2 = *(const float4*)(s1 + 16); float4 y3 = *(const float4*)(s1 + 20);
      a0[0]=x0.x; a0[1]=x0.y; a0[2]=x0.z; a0[3]=x0.w; a0[4]=x1.x; a0[5]=x1.y; a0[6]=x1.z; a0[7]=x1.w;
      a0[8]=x2.x; a0[9]=x2.y; a0[10]=x2.z; a0[11]=x2.w; a0[12]=x3.x; a0[13]=x3.y; a0[14]=x3.z; a0[15]=x3.w;
      a1[0]=y0.x; a1[1]=y0.y; a1[2]=y0.z; a1[3]=y0.w; a1[4]=y1.x; a1[5]=y1.y; a1[6]=y1.z; a1[7]=y1.w;
      a1[8]=y2.x; a1[9]=y2.y; a1[10]=y2.z; a1[11]=y2.w; a1[12]=y3.x; a1[13]=y3.y; a1[14]=y3.z; a1[15]=y3.w;
    } else {
#pragma unroll
      for (int e = 0; e < 16; ++e) {
        const long long off = (long long)(((e & 8) ? 16 : 0) + half * 8 + (e & 7)) * p.ak0;
        a0[e] = Ap0[off]; a1[e] = Ap1[off];
      }
    }
    if (BC) {
      const float* s0 = Bp0 + half * 8;
      const float* s1 = Bp1 + half * 8;
      float4 x0 = *(const float4*)(s0);      float4 x1 = *(const float4*)(s0 + 4);
      float4 x2 = *(const float4*)(s0 + 16); float4 x3 = *(const float4*)(s0 + 20);
      float4 y0 = *(const float4*)(s1);      float4 y1 = *(const float4*)(s1 + 4);
      float4 y2 = *(const float4*)(s1 + 16); float4 y3 = *(const float4*)(s1 + 20);
      v0[0]=x0.x; v0[1]=x0.y; v0[2]=x0.z; v0[3]=x0.w; v0[4]=x1.x; v0[5]=x1.y; v0[6]=x1.z; v0[7]=x1.w;
      v0[8]=x2.x; v0[9]=x2.y; v0[10]=x2.z; v0[11]=x2.w; v0[12]=x3.x; v0[13]=x3.y; v0[14]=x3.z; v0[15]=x3.w;
      v1[0]=y0.x; v1[1]=y0.y; v1[2]=y0.z; v1[3]=y0.w; v1[4]=y1.x; v1[5]=y1.y; v1[6]=y1.z; v1[7]=y1.w;
      v1[8]=y2.x; v1[9]=y2.y; v1[10]=y2.z; v1[11]=y2.w; v1[12]=y3.x; v1[13]=y3.y; v1[14]=y3.z; v1[15]=y3.w;
    } else {
#pragma unroll
      for (int e = 0; e < 16; ++e) {
        const long long off = (long long)(((e & 8) ? 16 : 0) + half * 8 + (e & 7)) * p.bk0;
        v0[e] = Bp0[off]; v1[e] = Bp1[off];
      }
    }
    bf16x16 fa0, fa1, fb0, fb1;
#pragma unroll
    for (int e = 0; e < 16; ++e) {
      fa0[e] = (__bf16)a0[e]; fa1[e] = (__bf16)a1[e];
      fb0[e] = (__bf16)v0[e]; fb1[e] = (__bf16)v1[e];
    }
    acc00 = wmma_bf16(fa0, fb0, acc00);
    acc01 = wmma_bf16(fa0, fb1, acc01);
    acc10 = wmma_bf16(fa1, fb0, acc10);
    acc11 = wmma_bf16(fa1, fb1, acc11);
    Ap0 += p.astep; Ap1 += p.astep; Bp0 += p.bstep; Bp1 += p.bstep;
  }

  float*       C = p.C + (long long)b0 * p.cb0 + (long long)b1 * p.cb1;
  const float* E = p.emul  ? p.emul  + (long long)b0 * p.cb0 + (long long)b1 * p.cb1 : nullptr;
  const float* R = p.resid ? p.resid + (long long)b0 * p.cb0 + (long long)b1 * p.cb1 : nullptr;
  const float* G = p.bias  ? p.bias  + (long long)b0 * p.gb0 + (long long)b1 * p.gb1 : nullptr;
  store_tile<NB>(p, C, E, R, G, acc00, m0,      nB0, half);
  store_tile<NB>(p, C, E, R, G, acc01, m0,      nB1, half);
  store_tile<NB>(p, C, E, R, G, acc10, m0 + 16, nB0, half);
  store_tile<NB>(p, C, E, R, G, acc11, m0 + 16, nB1, half);
}

// LayerNorm over last dim C (blockDim == C, one row per block, LDS reduce)
__global__ void ln_kernel(const float* __restrict__ x, const float* __restrict__ g,
                          const float* __restrict__ b, float* __restrict__ y, int C) {
  extern __shared__ float sm[];
  const long long row = blockIdx.x;
  const int t = threadIdx.x;
  const float v = x[row * C + t];
  sm[t] = v; __syncthreads();
  for (int s = blockDim.x >> 1; s > 0; s >>= 1) { if (t < s) sm[t] += sm[t + s]; __syncthreads(); }
  const float mean = sm[0] / (float)C; __syncthreads();
  const float d = v - mean;
  sm[t] = d * d; __syncthreads();
  for (int s = blockDim.x >> 1; s > 0; s >>= 1) { if (t < s) sm[t] += sm[t + s]; __syncthreads(); }
  const float var = sm[0] / (float)C;
  y[row * C + t] = d * rsqrtf(var + 1e-5f) * g[t] + b[t];
}

// Row-wise softmax, one wave (32 lanes) per row, L <= 192
__global__ void softmax_kernel(float* __restrict__ x, int L, long long rows) {
  const int wave = threadIdx.x >> 5, lane = threadIdx.x & 31;
  const long long row = (long long)blockIdx.x * 4 + wave;
  if (row >= rows) return;
  float* p = x + row * L;
  float mx = -1e30f;
  for (int j = lane; j < L; j += 32) mx = fmaxf(mx, p[j]);
  for (int o = 16; o > 0; o >>= 1) mx = fmaxf(mx, __shfl_xor(mx, o, 32));
  float s = 0.f;
  for (int j = lane; j < L; j += 32) { float e = __expf(p[j] - mx); p[j] = e; s += e; }
  for (int o = 16; o > 0; o >>= 1) s += __shfl_xor(s, o, 32);
  const float inv = 1.f / s;
  for (int j = lane; j < L; j += 32) p[j] *= inv;
}

// ---------------------------------------------------------------------------
// Host-side orchestration
// ---------------------------------------------------------------------------
static inline GemmParams GP() {
  GemmParams p{};
  p.scale = 1.f; p.biasScale = 1.f;
  return p;
}
static inline void launch(GemmParams p, hipStream_t st, int nbz) {
  if (p.astep == 0) p.astep = 32 * p.ak0;
  if (p.bstep == 0) p.bstep = 32 * p.bk0;
  dim3 grid((p.N + 31) / 32, (p.M + 31) / 32, nbz);
  const bool ac = (p.ak0 == 1), bc = (p.bk0 == 1), nb = (p.N % 32) != 0;
  if (nb)            wmma_gemm<true,  false, true ><<<grid, dim3(32), 0, st>>>(p);
  else if (ac && bc) wmma_gemm<true,  true,  false><<<grid, dim3(32), 0, st>>>(p);
  else if (ac)       wmma_gemm<true,  false, false><<<grid, dim3(32), 0, st>>>(p);
  else               wmma_gemm<false, false, false><<<grid, dim3(32), 0, st>>>(p);
}
// token-major GEMM: C[M,N] = epi(X[M,K] @ W[K,N])
static void gemmTW(hipStream_t st, const float* X, const float* W, const float* bias,
                   float* C, const float* emul, const float* resid,
                   int M, int K, int N, int act, float scale) {
  GemmParams p = GP();
  p.A = X; p.asr = K; p.ak0 = 1;
  p.B = W; p.bk0 = N; p.bn0 = 1;
  p.C = C; p.csr = N; p.cn0 = 1;
  p.bias = bias; p.gsc = 1;
  p.emul = emul; p.resid = resid;
  p.M = M; p.N = N; p.K = K; p.act = act; p.scale = scale;
  launch(p, st, 1);
}
static void run_ln(hipStream_t st, const float* x, const float* g, const float* b,
                   float* y, long long rows, int C) {
  ln_kernel<<<dim3((unsigned)rows), dim3(C), C * sizeof(float), st>>>(x, g, b, y, C);
}

struct AttnW { const float *wq, *wk, *wv, *wgate, *bgate, *wout, *bout; };

// Generic attention. Token row for (batch bi, position s) = bi*tb + s*ts.
//   row attn:  Bn=64 (m),  L=192, tb=192, ts=1
//   col attn:  Bn=192 (i), L=64,  tb=1,   ts=192
//   tri start: Bn=192 (i), L=192, tb=192, ts=1
//   tri end:   Bn=192 (j), L=192, tb=1,   ts=192
static void run_attention(hipStream_t st, const float* xln, int T, int inc, int Bn, int L,
                          long long tb, long long ts, const AttnW& w,
                          const float* biasPtr, long long b_sr, long long b_sc, long long b_b1,
                          float* q, float* k, float* v, float* o, float* aff,
                          float* Cout, const float* resid) {
  gemmTW(st, xln, w.wq, nullptr, q, nullptr, nullptr, T, inc, 256, 0, 1.f);
  gemmTW(st, xln, w.wk, nullptr, k, nullptr, nullptr, T, inc, 256, 0, 1.f);
  gemmTW(st, xln, w.wv, nullptr, v, nullptr, nullptr, T, inc, 256, 0, 1.f);
  // aff[(bi,h)][i,j] = (q_i . k_j)/sqrt(32) + bias
  GemmParams p = GP();
  p.A = q;   p.asr = ts * 256; p.ak0 = 1;        p.ab0 = tb * 256; p.ab1 = 32;
  p.B = k;   p.bk0 = 1;        p.bn0 = ts * 256; p.bb0 = tb * 256; p.bb1 = 32;
  p.C = aff; p.csr = L;        p.cn0 = 1;        p.cb0 = 8LL * L * L; p.cb1 = (long long)L * L;
  p.bias = biasPtr; p.gsr = b_sr; p.gsc = b_sc; p.gb0 = 0; p.gb1 = b_b1;
  p.M = L; p.N = L; p.K = 32; p.nbs = 3; p.scale = 0.17677669529663687f; // 1/sqrt(32)
  launch(p, st, Bn * 8);
  long long rows = (long long)Bn * 8 * L;
  softmax_kernel<<<dim3((unsigned)((rows + 3) / 4)), dim3(128), 0, st>>>(aff, L, rows);
  // o[(bi,h)][q,c] = sum_k aff[q,k] * v[k,c]  -> stored token-major
  p = GP();
  p.A = aff; p.asr = L;        p.ak0 = 1; p.ab0 = 8LL * L * L; p.ab1 = (long long)L * L;
  p.B = v;   p.bk0 = ts * 256; p.bn0 = 1; p.bb0 = tb * 256;    p.bb1 = 32;
  p.C = o;   p.csr = ts * 256; p.cn0 = 1; p.cb0 = tb * 256;    p.cb1 = 32;
  p.M = L; p.N = 32; p.K = L; p.nbs = 3;
  launch(p, st, Bn * 8);
  // o *= sigmoid(x @ wgate + bgate)  (fused via emul)
  gemmTW(st, xln, w.wgate, w.bgate, o, /*emul=*/o, nullptr, T, inc, 256, 2, 1.f);
  // out = o @ wout + bout + resid
  gemmTW(st, o, w.wout, w.bout, Cout, nullptr, resid, T, 256, inc, 0, 1.f);
}

static void run_biasproj(hipStream_t st, const float* pln, const float* wbias, float* biasb) {
  GemmParams p = GP();                 // Borig[h,i,j] stored h-major: h*36864 + i*192 + j
  p.A = pln;   p.asr = 128; p.ak0 = 1;
  p.B = wbias; p.bk0 = 8;   p.bn0 = 1;
  p.C = biasb; p.csr = 1;   p.cn0 = 36864;
  p.M = 36864; p.N = 8; p.K = 128;
  launch(p, st, 1);
}

static void run_trimult(hipStream_t st, bool ingoing, const float* const* prm,
                        float* pairAcc, float* pln, float* t1, float* t2, float* t3, float* t4) {
  // prm: ln1_g,ln1_b,ln2_g,ln2_b,wi,bi,wis,bis,wj,bj,wjs,bjs,wp,bp,wg,bg
  run_ln(st, pairAcc, prm[0], prm[1], pln, 36864, 128);
  gemmTW(st, pln, prm[4],  prm[5],  t1, nullptr, nullptr, 36864, 128, 128, 0, 1.f);
  gemmTW(st, pln, prm[6],  prm[7],  t1, t1,      nullptr, 36864, 128, 128, 2, 1.f);
  gemmTW(st, pln, prm[8],  prm[9],  t2, nullptr, nullptr, 36864, 128, 128, 0, 1.f);
  gemmTW(st, pln, prm[10], prm[11], t2, t2,      nullptr, 36864, 128, 128, 2, 1.f);
  // batched over channel c (128): out: o[i,j,c]=sum_k t1[i,k,c]*t2[j,k,c]
  //                               in:  o[i,j,c]=sum_k t1[k,j,c]*t2[k,i,c]
  GemmParams p = GP();
  if (!ingoing) {
    p.A = t1; p.asr = 24576; p.ak0 = 128;   p.ab0 = 1;
    p.B = t2; p.bk0 = 128;   p.bn0 = 24576; p.bb0 = 1;
  } else {
    p.A = t2; p.asr = 128;   p.ak0 = 24576; p.ab0 = 1;
    p.B = t1; p.bk0 = 24576; p.bn0 = 128;   p.bb0 = 1;
  }
  p.C = t3; p.csr = 24576; p.cn0 = 128; p.cb0 = 1;
  p.M = 192; p.N = 192; p.K = 192;
  launch(p, st, 128);
  run_ln(st, t3, prm[2], prm[3], t1, 36864, 128);                      // oln (reuse t1)
  gemmTW(st, pln, prm[14], prm[15], t4, nullptr, nullptr, 36864, 128, 128, 2, 1.f); // gate
  gemmTW(st, t1, prm[12], prm[13], pairAcc, t4, pairAcc, 36864, 128, 128, 0, 1.f);  // (oln@wp+bp)*g + pair
}

static void run_transition(hipStream_t st, float* acc, long long rows, int C, int NTf,
                           const float* lng, const float* lnb, const float* w1, const float* b1,
                           const float* w2, const float* b2, float* xln, float* h) {
  run_ln(st, acc, lng, lnb, xln, rows, C);
  gemmTW(st, xln, w1, b1, h, nullptr, nullptr, (int)rows, C, C * NTf, 1, 1.f);
  gemmTW(st, h, w2, b2, acc, nullptr, acc, (int)rows, C * NTf, C, 0, 1.f);
}

extern "C" void kernel_launch(void* const* d_in, const int* in_sizes, int n_in,
                              void* d_out, int out_size, void* d_ws, size_t ws_size,
                              hipStream_t stream) {
  (void)in_sizes; (void)n_in; (void)out_size; (void)ws_size;
  auto in = [&](int i) { return (const float*)d_in[i]; };
  float* ws = (float*)d_ws;
  // workspace layout (floats); total 122,191,872 f = ~489 MB
  float* q     = ws;
  float* k     = ws + 9437184;
  float* v     = ws + 18874368;
  float* o     = ws + 28311552;
  float* xln   = ws + 37748736;   // r1d LN   [12288,256]
  float* pln   = ws + 40894464;   // pair LN  [36864,128]
  float* biasb = ws + 45613056;   // [8,192,192]
  float* big   = ws + 45907968;   // aff / OPM outer / transition hidden (56,623,104 f)
  float* t1    = ws + 102531072;
  float* t2    = t1 + 4718592;
  float* t3    = t2 + 4718592;
  float* t4    = t3 + 4718592;
  float* opa   = t4 + 4718592;    // [64,192,32]
  float* opc   = opa + 393216;

  const float* r1d_in  = in(0);
  const float* pair_in = in(1);
  float* outR = (float*)d_out;          // r1d result [1,64,192,256]
  float* outP = outR + 3145728;         // pair result [1,192,192,128]

  // ===== row attention (reads d_in r1d + d_in pair) =====
  run_ln(stream, r1d_in, in(2), in(3), xln, 12288, 256);
  run_ln(stream, pair_in, in(12), in(13), pln, 36864, 128);
  run_biasproj(stream, pln, in(11), biasb);
  { AttnW w{in(4), in(5), in(6), in(7), in(8), in(9), in(10)};
    run_attention(stream, xln, 12288, 256, 64, 192, 192, 1, w,
                  biasb, 192, 1, 36864, q, k, v, o, big, outR, r1d_in); }

  // ===== column attention =====
  run_ln(stream, outR, in(14), in(15), xln, 12288, 256);
  { AttnW w{in(16), in(17), in(18), in(19), in(20), in(21), in(22)};
    run_attention(stream, xln, 12288, 256, 192, 64, 1, 192, w,
                  nullptr, 0, 0, 0, q, k, v, o, big, outR, outR); }

  // ===== MSA transition =====
  run_transition(stream, outR, 12288, 256, 4, in(23), in(24), in(25), in(26), in(27), in(28), xln, big);

  // ===== outer product mean =====
  run_ln(stream, outR, in(29), in(30), xln, 12288, 256);
  gemmTW(stream, xln, in(31), in(32), opa, nullptr, nullptr, 12288, 256, 32, 0, 1.f);
  gemmTW(stream, xln, in(33), in(34), opc, nullptr, nullptr, 12288, 256, 32, 0, 1.f);
  { // stage 1: T[(i,x),(j,y)] = sum_m a[m,i,x]*c[m,j,y]; rows r=i*32+x, cols n=j*32+y
    GemmParams p = GP();
    p.A = opa; p.asr = 1;    p.ak0 = 6144;
    p.B = opc; p.bk0 = 6144; p.bn0 = 1;
    p.C = big; p.csr = 6144; p.cn0 = 1;
    p.M = 6144; p.N = 6144; p.K = 64;
    launch(p, stream, 1);
  }
  { // stage 2 (batched over j): pair[i,j,:] = (T[(i,:),(j,:)] @ wf + bf)/64.001 + pair_in
    GemmParams p = GP();
    p.A = big;    p.asr = 196608; p.ak0 = 1;   p.astep = 6144; p.ab0 = 32;
    p.B = in(35); p.bk0 = 128;    p.bn0 = 1;                    // bstep defaults to 32*128
    p.C = outP;   p.csr = 24576;  p.cn0 = 1;   p.cb0 = 128;
    p.bias = in(36); p.gsc = 1;
    p.resid = pair_in;
    p.M = 192; p.N = 128; p.K = 1024;
    p.scale = 1.f / 64.001f; p.biasScale = 1.f / 64.001f;
    launch(p, stream, 192);
  }

  // ===== triangle multiplications =====
  { const float* prm[16]; for (int i = 0; i < 16; ++i) prm[i] = in(37 + i);
    run_trimult(stream, false, prm, outP, pln, t1, t2, t3, t4); }
  { const float* prm[16]; for (int i = 0; i < 16; ++i) prm[i] = in(53 + i);
    run_trimult(stream, true, prm, outP, pln, t1, t2, t3, t4); }

  // ===== triangle attention (starting node) =====
  run_ln(stream, outP, in(69), in(70), pln, 36864, 128);
  run_biasproj(stream, pln, in(78), biasb);
  { AttnW w{in(71), in(72), in(73), in(74), in(75), in(76), in(77)};
    run_attention(stream, pln, 36864, 128, 192, 192, 192, 1, w,
                  biasb, 192, 1, 36864, q, k, v, o, big, outP, outP); }

  // ===== triangle attention (ending node): all views via stride swap =====
  run_ln(stream, outP, in(79), in(80), pln, 36864, 128);
  run_biasproj(stream, pln, in(88), biasb);
  { AttnW w{in(81), in(82), in(83), in(84), in(85), in(86), in(87)};
    run_attention(stream, pln, 36864, 128, 192, 192, 1, 192, w,
                  biasb, 1, 192, 36864, q, k, v, o, big, outP, outP); }

  // ===== pair transition =====
  run_transition(stream, outP, 36864, 128, 4, in(89), in(90), in(91), in(92), in(93), in(94), pln, big);
}
